// MSA_Row_Att_41583873360510
// MI455X (gfx1250) — matrix operand order repaired and checked by reading, not compiled, not run on hardware
//
#include <hip/hip_runtime.h>
#include <hip/hip_bf16.h>
#include <math.h>
#include <stdint.h>

// ---------------------------------------------------------------------------
// MSA row attention for gfx1250 (CDNA5, wave32, WMMA 16x16x32 f16->f32)
// R=128, L=256, MSA_DIM=256, PAIR_DIM=128, H=8, D=32
// Async global->LDS staging (ASYNCcnt) with double-buffered tiles.
// ---------------------------------------------------------------------------

typedef __attribute__((ext_vector_type(16))) _Float16 v16h;
typedef __attribute__((ext_vector_type(8)))  float    v8f;
typedef __attribute__((ext_vector_type(4)))  int      i32x4;

typedef __attribute__((address_space(1))) i32x4 gi32x4;   // global int4
typedef __attribute__((address_space(3))) i32x4 li32x4;   // LDS int4

union AFrag {            // 8 VGPRs = 16 f16 = 32 bytes
    v16h  h;
    uint4 u4[2];
    _Float16 e[16];
};

__device__ __forceinline__ v8f v8f_zero() {
    v8f z;
#pragma unroll
    for (int i = 0; i < 8; ++i) z[i] = 0.0f;
    return z;
}

__device__ __forceinline__ v8f wmma_f16(const AFrag& a, const AFrag& b, v8f c) {
    return __builtin_amdgcn_wmma_f32_16x16x32_f16(
        /*neg_a=*/false, a.h, /*neg_b=*/false, b.h,
        /*c_mod=*/(short)0, c, /*reuse_a=*/false, /*reuse_b=*/false);
}

#if __has_builtin(__builtin_amdgcn_global_load_async_to_lds_b128)
#define HAVE_ASYNC_COPY 1
#else
#define HAVE_ASYNC_COPY 0
#endif

// 16-byte global -> LDS copy.  Async (ASYNCcnt) when the toolchain exposes
// GLOBAL_LOAD_ASYNC_TO_LDS_B128; otherwise a sync VGPR round-trip.
__device__ __forceinline__ void copy16_to_lds(const _Float16* g, _Float16* l) {
#if HAVE_ASYNC_COPY
    // generic->AS casts via integer: AS1 keeps the 64-bit value; AS3 is the
    // low 32 bits of the generic shared pointer (LDS aperture = addr[31:0]).
    __builtin_amdgcn_global_load_async_to_lds_b128(
        (gi32x4*)(uintptr_t)g,
        (li32x4*)(uint32_t)(uintptr_t)l,
        /*offset=*/0, /*cpol=*/0);
#else
    *(uint4*)l = *(const uint4*)g;
#endif
}

// Wait for this wave's outstanding async LDS copies (no-op in fallback).
__device__ __forceinline__ void async_copy_fence() {
#if HAVE_ASYNC_COPY
#if __has_builtin(__builtin_amdgcn_s_wait_asynccnt)
    __builtin_amdgcn_s_wait_asynccnt(0);
#else
    asm volatile("s_wait_asynccnt 0x0" ::: "memory");
#endif
#endif
}

static constexpr int R = 128;
static constexpr int L = 256;
static constexpr int MSA = 256;
static constexpr int PAIR = 128;
static constexpr int H = 8;
static constexpr int D = 32;

// ---------------------------------------------------------------------------
// Kernel 1: LayerNorm over last dim -> f16.  One wave per row.
// ---------------------------------------------------------------------------
__global__ __launch_bounds__(256)
void msa_ln_kernel(const float* __restrict__ x,
                   const float* __restrict__ gamma,
                   const float* __restrict__ beta,
                   _Float16* __restrict__ out, int cols)
{
    const int lane = threadIdx.x & 31;
    const int rib  = threadIdx.x >> 5;
    const long row = (long)blockIdx.x * (blockDim.x >> 5) + rib;
    const int per  = cols >> 5;            // 8 for MSA, 4 for PAIR

    const float* xp = x + row * (long)cols + lane * per;
    float vals[8];
    float s = 0.f, s2 = 0.f;
#pragma unroll
    for (int i = 0; i < 8; ++i) {
        if (i < per) {
            float v = xp[i];
            vals[i] = v; s += v; s2 += v * v;
        }
    }
#pragma unroll
    for (int off = 1; off < 32; off <<= 1) {
        s  += __shfl_xor(s,  off, 32);
        s2 += __shfl_xor(s2, off, 32);
    }
    const float mean = s / (float)cols;
    const float var  = s2 / (float)cols - mean * mean;
    const float inv  = rsqrtf(var + 1e-5f);

    _Float16* op = out + row * (long)cols + lane * per;
    const float* gp = gamma + lane * per;
    const float* bp = beta  + lane * per;
#pragma unroll
    for (int i = 0; i < 8; ++i) {
        if (i < per)
            op[i] = (_Float16)((vals[i] - mean) * inv * gp[i] + bp[i]);
    }
}

// ---------------------------------------------------------------------------
// Kernel 2: fused LayerNorm(pair) + z@Wb -> bias[H][L][L] (f32)
// ---------------------------------------------------------------------------
__global__ __launch_bounds__(256)
void msa_zbias_kernel(const float* __restrict__ pair,
                      const float* __restrict__ zg,
                      const float* __restrict__ zb,
                      const float* __restrict__ Wb,
                      float* __restrict__ bias)
{
    const int idx = blockIdx.x * blockDim.x + threadIdx.x;   // i*L + j
    const float* xp = pair + (size_t)idx * PAIR;

    float s = 0.f, s2 = 0.f;
#pragma unroll 4
    for (int k = 0; k < PAIR; ++k) { float v = xp[k]; s += v; s2 += v * v; }
    const float mean = s * (1.0f / PAIR);
    const float var  = s2 * (1.0f / PAIR) - mean * mean;
    const float inv  = rsqrtf(var + 1e-5f);

    float acc[H];
#pragma unroll
    for (int h = 0; h < H; ++h) acc[h] = 0.f;
#pragma unroll 2
    for (int k = 0; k < PAIR; ++k) {
        const float zn = (xp[k] - mean) * inv * zg[k] + zb[k];
        const float* wr = Wb + k * H;
#pragma unroll
        for (int h = 0; h < H; ++h) acc[h] += zn * wr[h];
    }
#pragma unroll
    for (int h = 0; h < H; ++h)
        bias[(size_t)h * (L * L) + idx] = acc[h];
}

// ---------------------------------------------------------------------------
// Kernel 3: weight convert + transpose: WT16[n*K + k] = (f16)W[k*N + n]
// ---------------------------------------------------------------------------
__global__ __launch_bounds__(256)
void msa_wconvT_kernel(const float* __restrict__ W, _Float16* __restrict__ WT,
                       int K, int N)
{
    const int n = blockIdx.x;
    const int k = threadIdx.x;
    WT[(size_t)n * K + k] = (_Float16)W[(size_t)k * N + n];
}

// ---------------------------------------------------------------------------
// Kernel 4: tiled WMMA GEMM  out[M,256] = A16[M,256] @ W (given transposed)
// Block = 128 threads (4 waves); tile 64(M) x 64(N); K steps of 32.
// Double-buffered LDS tiles filled with async global->LDS copies.
// mode 0: f16 out (k,v) | 1: f16 *scale (q) | 2: f16 sigmoid(x+bg) (gate)
// mode 3: f32 out = x + bo (final projection)
// ---------------------------------------------------------------------------
__global__ __launch_bounds__(128)
void msa_gemm_kernel(const _Float16* __restrict__ A,
                     const _Float16* __restrict__ WT,
                     _Float16* __restrict__ out16,
                     float* __restrict__ out32,
                     const float* __restrict__ bvec,
                     int mode, float scale)
{
    __shared__ _Float16 ldsA[2][64 * 32];
    __shared__ _Float16 ldsB[2][64 * 32];

    const int t    = threadIdx.x;
    const int wv   = t >> 5;
    const int lane = t & 31;
    const int ln   = lane & 15;
    const int hi   = lane >> 4;
    const int m0   = blockIdx.x * 64;
    const int n0   = blockIdx.y * 64;

    const int ml = t >> 1;            // 0..63
    const int ko = (t & 1) * 16;      // 0 or 16

    // stage one 64x32 A tile + 64x32 B tile into buffer `buf` (async)
    auto stage = [&](int buf, int kk) {
        const _Float16* ga = A  + (size_t)(m0 + ml) * MSA + kk + ko;
        const _Float16* gb = WT + (size_t)(n0 + ml) * MSA + kk + ko;
        _Float16* la = &ldsA[buf][ml * 32 + ko];
        _Float16* lb = &ldsB[buf][ml * 32 + ko];
        copy16_to_lds(ga,     la);
        copy16_to_lds(ga + 8, la + 8);
        copy16_to_lds(gb,     lb);
        copy16_to_lds(gb + 8, lb + 8);
    };

    v8f acc[4];
#pragma unroll
    for (int c = 0; c < 4; ++c) acc[c] = v8f_zero();

    stage(0, 0);
    int cur = 0;

    for (int kk = 0; kk < MSA; kk += 32) {
        async_copy_fence();
        __syncthreads();                     // buf[cur] complete + visible
        if (kk + 32 < MSA) stage(cur ^ 1, kk + 32);   // overlap with compute

        // A fragment: lane ln = row (wv*16+ln); K run [16*hi,+16)
        AFrag fa;
        const _Float16* ap = &ldsA[cur][(wv * 16 + ln) * 32];
        fa.u4[0] = *(const uint4*)(ap + 16 * hi);
        fa.u4[1] = *(const uint4*)(ap + 16 * hi + 8);

#pragma unroll
        for (int c = 0; c < 4; ++c) {
            AFrag fb;   // B fragment: lane ln = column, K run [16*hi,+16)
            const _Float16* bp = &ldsB[cur][(c * 16 + ln) * 32 + 16 * hi];
            fb.u4[0] = ((const uint4*)bp)[0];
            fb.u4[1] = ((const uint4*)bp)[1];
            acc[c] = wmma_f16(fa, fb, acc[c]);
        }
        cur ^= 1;
    }

    // epilogue: C layout row = v + 8*hi, col = ln (per 16-col block)
#pragma unroll
    for (int c = 0; c < 4; ++c) {
        const int col = n0 + c * 16 + ln;
#pragma unroll
        for (int v = 0; v < 8; ++v) {
            const int row = m0 + wv * 16 + v + 8 * hi;
            float x = acc[c][v];
            if (mode == 0) {
                out16[(size_t)row * MSA + col] = (_Float16)x;
            } else if (mode == 1) {
                out16[(size_t)row * MSA + col] = (_Float16)(x * scale);
            } else if (mode == 2) {
                float g = 1.0f / (1.0f + __expf(-(x + bvec[col])));
                out16[(size_t)row * MSA + col] = (_Float16)g;
            } else {
                out32[(size_t)row * MSA + col] = x + bvec[col];
            }
        }
    }
}

// ---------------------------------------------------------------------------
// Kernel 5: flash attention per (r, h).  Block = 128 threads = 4 waves;
// each wave owns one 16-row i-tile; 8 iterations over 32-column j-blocks,
// double-buffered async K/V staging.
//   S(16x32)  : 2x WMMA (K = D = 32), + pair bias, online softmax
//   O += P*V  : P(16x32 f16 via same-wave LDS round-trip) x V -> 2x WMMA
// ---------------------------------------------------------------------------
__global__ __launch_bounds__(128)
void msa_attn_kernel(const _Float16* __restrict__ q16,
                     const _Float16* __restrict__ k16,
                     const _Float16* __restrict__ v16,
                     const _Float16* __restrict__ g16,
                     const float* __restrict__ bias,
                     _Float16* __restrict__ ctx16)
{
    __shared__ _Float16 ldsK[2][32 * 32];    // [j_local][d]
    __shared__ _Float16 ldsV[2][32 * 32];    // [j_local][d]
    __shared__ _Float16 ldsP[4][16 * 32];    // per-wave P scratch

    const int t    = threadIdx.x;
    const int wv   = t >> 5;
    const int lane = t & 31;
    const int ln   = lane & 15;
    const int hi   = lane >> 4;

    const int b   = blockIdx.x;          // R * H * 4 blocks
    const int r   = b >> 5;              // / (H*4)
    const int rem = b & 31;
    const int h   = rem >> 2;
    const int i0  = ((rem & 3) * 4 + wv) * 16;

    const int jl = t >> 2;               // 0..31
    const int d0 = (t & 3) * 8;          // 0,8,16,24

    auto stageKV = [&](int buf, int j0) {
        const size_t rowb = ((size_t)(r * L + j0 + jl)) * MSA + h * D + d0;
        copy16_to_lds(k16 + rowb, &ldsK[buf][jl * 32 + d0]);
        copy16_to_lds(v16 + rowb, &ldsV[buf][jl * 32 + d0]);
    };

    // loop-invariant Q fragment (A layout), rows i0+ln, K = d
    AFrag fq;
    {
        const _Float16* qp = q16 + ((size_t)(r * L + i0 + ln)) * MSA + h * D;
        fq.u4[0] = *(const uint4*)(qp + 16 * hi);
        fq.u4[1] = *(const uint4*)(qp + 16 * hi + 8);
    }

    float rmax[8], lsum[8];
#pragma unroll
    for (int v = 0; v < 8; ++v) { rmax[v] = -1e30f; lsum[v] = 0.f; }
    v8f o0 = v8f_zero(), o1 = v8f_zero();

    const float* biash = bias + (size_t)h * (L * L);

    stageKV(0, 0);
    int cur = 0;

    for (int j0 = 0; j0 < L; j0 += 32) {
        async_copy_fence();
        __syncthreads();                      // buf[cur] complete + visible
        if (j0 + 32 < L) stageKV(cur ^ 1, j0 + 32);

        // S = q . k^T : B columns are rows of k (contiguous in d)
        v8f s0 = v8f_zero(), s1 = v8f_zero();
        {
            AFrag fk;
            const _Float16* kp = &ldsK[cur][ln * 32 + 16 * hi];
            fk.u4[0] = ((const uint4*)kp)[0];
            fk.u4[1] = ((const uint4*)kp)[1];
            s0 = wmma_f16(fq, fk, s0);
            const _Float16* kp2 = &ldsK[cur][(16 + ln) * 32 + 16 * hi];
            fk.u4[0] = ((const uint4*)kp2)[0];
            fk.u4[1] = ((const uint4*)kp2)[1];
            s1 = wmma_f16(fq, fk, s1);
        }

        // + pair bias (rows v+8*hi live in this lane group)
#pragma unroll
        for (int v = 0; v < 8; ++v) {
            const int irow = i0 + v + 8 * hi;
            s0[v] += biash[(size_t)irow * L + j0 + ln];
            s1[v] += biash[(size_t)irow * L + j0 + 16 + ln];
        }
        // online softmax: 16-lane row reductions
#pragma unroll
        for (int v = 0; v < 8; ++v) {
            float tmax = fmaxf(s0[v], s1[v]);
#pragma unroll
            for (int off = 1; off < 16; off <<= 1)
                tmax = fmaxf(tmax, __shfl_xor(tmax, off, 32));
            const float mnew = fmaxf(rmax[v], tmax);
            const float corr = __expf(rmax[v] - mnew);
            const float e0 = __expf(s0[v] - mnew);
            const float e1 = __expf(s1[v] - mnew);
            s0[v] = e0; s1[v] = e1;
            float ps = e0 + e1;
#pragma unroll
            for (int off = 1; off < 16; off <<= 1)
                ps += __shfl_xor(ps, off, 32);
            lsum[v] = lsum[v] * corr + ps;
            o0[v] *= corr; o1[v] *= corr;
            rmax[v] = mnew;
        }

        // P -> LDS (C layout -> row-major), then back as A fragment.
        // Same-wave LDS ops are in-order on CDNA5 (DScnt): no barrier needed.
        _Float16* pb = ldsP[wv];
#pragma unroll
        for (int v = 0; v < 8; ++v) {
            pb[(v + 8 * hi) * 32 + ln]      = (_Float16)s0[v];
            pb[(v + 8 * hi) * 32 + 16 + ln] = (_Float16)s1[v];
        }
        AFrag fp;
        {
            const _Float16* pp = pb + ln * 32;
            fp.u4[0] = *(const uint4*)(pp + 16 * hi);
            fp.u4[1] = *(const uint4*)(pp + 16 * hi + 8);
        }

        // O += P * V ; B column d gathered with strided LDS reads (V is
        // staged row-major [j][d]); K run = j = 16*hi + e.
        {
            AFrag fv0, fv1;
#pragma unroll
            for (int i = 0; i < 16; ++i) {
                const int jrow = 16 * hi + i;
                fv0.e[i] = ldsV[cur][jrow * 32 + ln];
                fv1.e[i] = ldsV[cur][jrow * 32 + 16 + ln];
            }
            o0 = wmma_f16(fp, fv0, o0);
            o1 = wmma_f16(fp, fv1, o1);
        }
        cur ^= 1;
    }

    // normalize, gate, write ctx (f16)
#pragma unroll
    for (int v = 0; v < 8; ++v) {
        const int irow = i0 + v + 8 * hi;
        const float inv = 1.0f / lsum[v];
        const size_t base = ((size_t)(r * L + irow)) * MSA + h * D;
        const float g0 = (float)g16[base + ln];
        const float g1 = (float)g16[base + 16 + ln];
        ctx16[base + ln]      = (_Float16)(o0[v] * inv * g0);
        ctx16[base + 16 + ln] = (_Float16)(o1[v] * inv * g1);
    }
}

// ---------------------------------------------------------------------------
// Host launcher
// ---------------------------------------------------------------------------
extern "C" void kernel_launch(void* const* d_in, const int* in_sizes, int n_in,
                              void* d_out, int out_size, void* d_ws, size_t ws_size,
                              hipStream_t stream)
{
    (void)in_sizes; (void)n_in; (void)out_size; (void)ws_size;

    const float* msa   = (const float*)d_in[0];
    const float* pair  = (const float*)d_in[1];
    const float* m_g   = (const float*)d_in[2];
    const float* m_b   = (const float*)d_in[3];
    const float* z_g   = (const float*)d_in[4];
    const float* z_b   = (const float*)d_in[5];
    const float* Wb    = (const float*)d_in[6];
    const float* Wq    = (const float*)d_in[7];
    const float* Wk    = (const float*)d_in[8];
    const float* Wv    = (const float*)d_in[9];
    const float* Wg    = (const float*)d_in[10];
    const float* bg    = (const float*)d_in[11];
    const float* Wo    = (const float*)d_in[12];
    const float* bo    = (const float*)d_in[13];
    float* out = (float*)d_out;

    char* ws = (char*)d_ws;
    const size_t SZ16 = (size_t)R * L * MSA * sizeof(_Float16);   // 16 MB
    _Float16* m16   = (_Float16*)(ws);
    _Float16* q16   = (_Float16*)(ws + 1 * SZ16);
    _Float16* k16   = (_Float16*)(ws + 2 * SZ16);
    _Float16* v16   = (_Float16*)(ws + 3 * SZ16);
    _Float16* g16   = (_Float16*)(ws + 4 * SZ16);
    _Float16* ctx16 = (_Float16*)(ws + 5 * SZ16);
    float*    biasb = (float*)(ws + 6 * SZ16);                    // H*L*L f32
    char* wbase = ws + 6 * SZ16 + (size_t)H * L * L * sizeof(float);
    const size_t WSZ = (size_t)MSA * MSA * sizeof(_Float16);      // 128 KB
    _Float16* wqT = (_Float16*)(wbase + 0 * WSZ);
    _Float16* wkT = (_Float16*)(wbase + 1 * WSZ);
    _Float16* wvT = (_Float16*)(wbase + 2 * WSZ);
    _Float16* wgT = (_Float16*)(wbase + 3 * WSZ);
    _Float16* woT = (_Float16*)(wbase + 4 * WSZ);

    const int M = R * L;                       // 32768 rows

    // 1) LayerNorm of MSA embedding -> f16
    msa_ln_kernel<<<M / 8, 256, 0, stream>>>(msa, m_g, m_b, m16, MSA);

    // 2) fused LayerNorm(pair) + z@Wb -> bias
    msa_zbias_kernel<<<(L * L) / 256, 256, 0, stream>>>(pair, z_g, z_b, Wb, biasb);

    // 3) weight convert+transpose to f16
    msa_wconvT_kernel<<<MSA, MSA, 0, stream>>>(Wq, wqT, MSA, MSA);
    msa_wconvT_kernel<<<MSA, MSA, 0, stream>>>(Wk, wkT, MSA, MSA);
    msa_wconvT_kernel<<<MSA, MSA, 0, stream>>>(Wv, wvT, MSA, MSA);
    msa_wconvT_kernel<<<MSA, MSA, 0, stream>>>(Wg, wgT, MSA, MSA);
    msa_wconvT_kernel<<<MSA, MSA, 0, stream>>>(Wo, woT, MSA, MSA);

    // 4) projections (WMMA GEMMs)
    dim3 ggrid(M / 64, MSA / 64);
    const float qscale = 0.17677669529663688f;   // 1/sqrt(32)
    msa_gemm_kernel<<<ggrid, 128, 0, stream>>>(m16, wqT, q16, nullptr, nullptr, 1, qscale);
    msa_gemm_kernel<<<ggrid, 128, 0, stream>>>(m16, wkT, k16, nullptr, nullptr, 0, 1.0f);
    msa_gemm_kernel<<<ggrid, 128, 0, stream>>>(m16, wvT, v16, nullptr, nullptr, 0, 1.0f);
    msa_gemm_kernel<<<ggrid, 128, 0, stream>>>(m16, wgT, g16, nullptr, bg,      2, 1.0f);

    // 5) flash attention (WMMA) -> ctx16
    msa_attn_kernel<<<R * H * 4, 128, 0, stream>>>(q16, k16, v16, g16, biasb, ctx16);

    // 6) output projection: ctx @ Wo + bo -> f32
    msa_gemm_kernel<<<ggrid, 128, 0, stream>>>(ctx16, woT, nullptr, out, bo, 3, 1.0f);
}
